// TitanMemory_13649406067431
// MI455X (gfx1250) — matrix operand order repaired and checked by reading, not compiled, not run on hardware
//
#include <hip/hip_runtime.h>
#include <hip/hip_bf16.h>
#include <math.h>

// Problem constants (B=8, S=4096, DIM=1024, M=1024)
#define DIMK  1024
#define MMEM  1024
#define NROWS 32768   // B*S

typedef __bf16 bf16;
typedef __attribute__((ext_vector_type(16))) __bf16 v16bf;
typedef __attribute__((ext_vector_type(8)))  __bf16 v8bf;
typedef __attribute__((ext_vector_type(4)))  __bf16 v4bf;
typedef __attribute__((ext_vector_type(8)))  float  v8f;
typedef __attribute__((ext_vector_type(4)))  float  v4f;

// ---- WMMA fragment loaders (wave32, 16x16x32 bf16 layouts per CDNA5 ISA §7.12.2) ----
// A (16x32, row-major source [16][ld]): lane(0-15)=row, per-lane K runs
//   [k0+8*half, +8) in VGPRs 0-3 and [k0+16+8*half, +8) in VGPRs 4-7.
static __device__ __forceinline__ v16bf load_a16(const bf16* base, int ld, int k0,
                                                 int lm, int half) {
  const bf16* p = base + lm * ld + k0 + half * 8;
  v8bf lo = *(const v8bf*)(p);
  v8bf hi = *(const v8bf*)(p + 16);
  v16bf r;
#pragma unroll
  for (int i = 0; i < 8; ++i) { r[i] = lo[i]; r[8 + i] = hi[i]; }
  return r;
}

// B (32x16): column n = lane%16 contiguous along K in source row (n0+lm) of [*][ld];
// per-lane K runs [k0+8*half, +8) and [k0+16+8*half, +8): two aligned 16-byte loads.
static __device__ __forceinline__ v16bf load_b16(const bf16* src, int ld, int n0,
                                                 int k0, int lm, int half) {
  const bf16* p = src + (size_t)(n0 + lm) * ld + k0 + half * 8;
  v8bf lo = *(const v8bf*)(p);
  v8bf hi = *(const v8bf*)(p + 16);
  v16bf r;
#pragma unroll
  for (int i = 0; i < 8; ++i) { r[i] = lo[i]; r[8 + i] = hi[i]; }
  return r;
}

static __device__ __forceinline__ v8f wmma_bf16(v16bf a, v16bf b, v8f c) {
  return __builtin_amdgcn_wmma_f32_16x16x32_bf16(false, a, false, b, (short)0, c,
                                                 false, false);
}

// ---- Pre-pass: fp32 -> bf16 conversions (W, mem, mem^T) ----
__global__ void titan_convert(const float* __restrict__ W, const float* __restrict__ mem,
                              bf16* __restrict__ Wb, bf16* __restrict__ memb,
                              bf16* __restrict__ memTb) {
  int i = blockIdx.x * blockDim.x + threadIdx.x;
  if (i < DIMK * DIMK) {
    Wb[i] = (bf16)W[i];
    float mv = mem[i];
    memb[i] = (bf16)mv;
    int r = i >> 10;         // mem row (m)
    int c = i & (DIMK - 1);  // mem col (d)
    memTb[(size_t)c * MMEM + r] = (bf16)mv;
  }
}

// ---- Fused q-proj + attention + context, flash-style, one 16-row tile per wave ----
// 4-wide N-blocking: one A fragment feeds 4 WMMAs => LDS A-traffic / 4.
__global__ __launch_bounds__(64) void titan_fused(
    const float* __restrict__ x, const float* __restrict__ bvec,
    const bf16* __restrict__ Wb, const bf16* __restrict__ memb,
    const bf16* __restrict__ memTb, float* __restrict__ out) {
  // Per-wave regions: xs = x-tile bf16, later reused for scores/probs; qs = q-tile bf16.
  __shared__ bf16  lds_s[2][16 * DIMK];   // 2 * 32 KB
  __shared__ bf16  lds_q[2][16 * DIMK];   // 2 * 32 KB
  __shared__ float lds_max[2][16];
  __shared__ float lds_inv[2][16];

  const int wave = threadIdx.x >> 5;
  const int lane = threadIdx.x & 31;
  const int half = lane >> 4;
  const int lm   = lane & 15;
  const int tile = blockIdx.x * 2 + wave;
  const long r0  = (long)tile * 16;

  bf16* xs = lds_s[wave];
  bf16* qs = lds_q[wave];

  // ---- Stage x tile (16x1024 f32, contiguous) into LDS as bf16 ----
  const float* xsrc = x + r0 * DIMK;
  for (int idx = lane * 4; idx < 16 * DIMK; idx += 32 * 4) {
    v4f v = *(const v4f*)(xsrc + idx);
    v4bf o;
    o[0] = (bf16)v[0]; o[1] = (bf16)v[1]; o[2] = (bf16)v[2]; o[3] = (bf16)v[3];
    *(v4bf*)(xs + idx) = o;
  }
  __syncthreads();

  // ---- Phase A: q = x @ W^T + b   (B[k][n] = W[n0+n][k], row n contiguous in K) ----
  for (int n0 = 0; n0 < DIMK; n0 += 64) {
    if (n0 + 64 < DIMK)
      __builtin_prefetch(Wb + (size_t)(n0 + 64 + lm) * DIMK, 0, 0);
    v8f acc[4];
#pragma unroll
    for (int j = 0; j < 4; ++j) acc[j] = (v8f){0.f,0.f,0.f,0.f,0.f,0.f,0.f,0.f};
#pragma unroll 2
    for (int k0 = 0; k0 < DIMK; k0 += 32) {
      v16bf a  = load_a16(xs, DIMK, k0, lm, half);
      v16bf b0 = load_b16(Wb, DIMK, n0 +  0, k0, lm, half);
      v16bf b1 = load_b16(Wb, DIMK, n0 + 16, k0, lm, half);
      v16bf b2 = load_b16(Wb, DIMK, n0 + 32, k0, lm, half);
      v16bf b3 = load_b16(Wb, DIMK, n0 + 48, k0, lm, half);
      acc[0] = wmma_bf16(a, b0, acc[0]);
      acc[1] = wmma_bf16(a, b1, acc[1]);
      acc[2] = wmma_bf16(a, b2, acc[2]);
      acc[3] = wmma_bf16(a, b3, acc[3]);
    }
#pragma unroll
    for (int j = 0; j < 4; ++j) {
      float bb = bvec[n0 + 16 * j + lm];
#pragma unroll
      for (int i = 0; i < 8; ++i)
        qs[(i + 8 * half) * DIMK + n0 + 16 * j + lm] = (bf16)(acc[j][i] + bb);
    }
  }

  // ---- Phase B1: scores = q @ mem^T with online softmax stats ----
  float rmax[8], rsum[8];
#pragma unroll
  for (int i = 0; i < 8; ++i) { rmax[i] = -3.0e38f; rsum[i] = 0.f; }

  for (int m0 = 0; m0 < MMEM; m0 += 64) {
    v8f acc[4];
#pragma unroll
    for (int j = 0; j < 4; ++j) acc[j] = (v8f){0.f,0.f,0.f,0.f,0.f,0.f,0.f,0.f};
#pragma unroll 2
    for (int k0 = 0; k0 < DIMK; k0 += 32) {
      v16bf a  = load_a16(qs, DIMK, k0, lm, half);
      v16bf b0 = load_b16(memb, DIMK, m0 +  0, k0, lm, half);
      v16bf b1 = load_b16(memb, DIMK, m0 + 16, k0, lm, half);
      v16bf b2 = load_b16(memb, DIMK, m0 + 32, k0, lm, half);
      v16bf b3 = load_b16(memb, DIMK, m0 + 48, k0, lm, half);
      acc[0] = wmma_bf16(a, b0, acc[0]);
      acc[1] = wmma_bf16(a, b1, acc[1]);
      acc[2] = wmma_bf16(a, b2, acc[2]);
      acc[3] = wmma_bf16(a, b3, acc[3]);
    }
#pragma unroll
    for (int j = 0; j < 4; ++j) {
#pragma unroll
      for (int i = 0; i < 8; ++i) {
        float v = acc[j][i];
        float tmax = v;                      // reduce over the 16 N-lanes (bits 0..3)
        for (int off = 1; off < 16; off <<= 1)
          tmax = fmaxf(tmax, __shfl_xor(tmax, off, 32));
        float nm = fmaxf(rmax[i], tmax);
        float e  = __expf(v - nm);
        float ts = e;
        for (int off = 1; off < 16; off <<= 1)
          ts += __shfl_xor(ts, off, 32);
        rsum[i] = rsum[i] * __expf(rmax[i] - nm) + ts;
        rmax[i] = nm;
        xs[(i + 8 * half) * MMEM + m0 + 16 * j + lm] = (bf16)v;   // raw score
      }
    }
  }
  if (lm == 0) {
#pragma unroll
    for (int i = 0; i < 8; ++i) {
      lds_max[wave][i + 8 * half] = rmax[i];
      lds_inv[wave][i + 8 * half] = 1.0f / rsum[i];
    }
  }
  __syncthreads();

  // ---- Normalize scores -> probabilities (bf16), in place in xs ----
  for (int idx = lane; idx < 16 * MMEM; idx += 32) {
    int row = idx >> 10;
    float sc = (float)xs[idx];
    float p = __expf(sc - lds_max[wave][row]) * lds_inv[wave][row];
    xs[idx] = (bf16)p;
  }
  __syncthreads();

  // ---- Phase B2: context = P @ mem   (B[k][n] = mem[m0+k][d0+n] = memT[d0+n][m0+k]) ----
  for (int d0 = 0; d0 < DIMK; d0 += 64) {
    if (d0 + 64 < DIMK)
      __builtin_prefetch(memTb + (size_t)(d0 + 64 + lm) * MMEM, 0, 0);
    v8f acc[4];
#pragma unroll
    for (int j = 0; j < 4; ++j) acc[j] = (v8f){0.f,0.f,0.f,0.f,0.f,0.f,0.f,0.f};
#pragma unroll 2
    for (int m0 = 0; m0 < MMEM; m0 += 32) {
      v16bf a  = load_a16(xs, MMEM, m0, lm, half);
      v16bf b0 = load_b16(memTb, MMEM, d0 +  0, m0, lm, half);
      v16bf b1 = load_b16(memTb, MMEM, d0 + 16, m0, lm, half);
      v16bf b2 = load_b16(memTb, MMEM, d0 + 32, m0, lm, half);
      v16bf b3 = load_b16(memTb, MMEM, d0 + 48, m0, lm, half);
      acc[0] = wmma_bf16(a, b0, acc[0]);
      acc[1] = wmma_bf16(a, b1, acc[1]);
      acc[2] = wmma_bf16(a, b2, acc[2]);
      acc[3] = wmma_bf16(a, b3, acc[3]);
    }
#pragma unroll
    for (int j = 0; j < 4; ++j)
#pragma unroll
      for (int i = 0; i < 8; ++i)
        out[(r0 + i + 8 * half) * DIMK + d0 + 16 * j + lm] = acc[j][i];
  }
}

extern "C" void kernel_launch(void* const* d_in, const int* in_sizes, int n_in,
                              void* d_out, int out_size, void* d_ws, size_t ws_size,
                              hipStream_t stream) {
  const float* x   = (const float*)d_in[0];  // [B,S,DIM]
  const float* W   = (const float*)d_in[1];  // [DIM,DIM]
  const float* b   = (const float*)d_in[2];  // [DIM]
  const float* mem = (const float*)d_in[3];  // [1,M,DIM]

  bf16* Wb    = (bf16*)d_ws;                 // 2 MB
  bf16* memb  = Wb + (size_t)DIMK * DIMK;    // 2 MB
  bf16* memTb = memb + (size_t)MMEM * DIMK;  // 2 MB  (total 6 MB workspace)

  titan_convert<<<(DIMK * DIMK + 255) / 256, 256, 0, stream>>>(W, mem, Wb, memb, memTb);

  // 2048 row-tiles of 16, 2 waves (tiles) per block of 64 threads -> 1024 blocks
  titan_fused<<<NROWS / 32, 64, 0, stream>>>(x, b, Wb, memb, memTb, (float*)d_out);
}
